// TokenShufflingMoE_13116830122700
// MI455X (gfx1250) — compile-verified, matrix operand order checked by reading, not run on hardware
//
#include <hip/hip_runtime.h>
#include <hip/hip_bf16.h>
#include <math.h>

// ---------------------------------------------------------------------------
// TokenShufflingMoE for MI455X (gfx1250, wave32, WMMA bf16 16x16x32)
// B=2, T=1024 -> N=2048 tokens, D=1024, E=8, H=1024, Hs=2048, top-1 routing.
// ---------------------------------------------------------------------------

typedef __bf16 bf16_t;
typedef __attribute__((ext_vector_type(16))) __bf16 v16bf;
typedef __attribute__((ext_vector_type(8)))  __bf16 v8bf;
typedef __attribute__((ext_vector_type(8)))  float  v8f;
typedef __attribute__((ext_vector_type(4)))  float  v4f;

#define WMMA_BF16(a, b, c) \
  __builtin_amdgcn_wmma_f32_16x16x32_bf16(false, (a), false, (b), (short)0, (c), false, false)

#define MOE_N   2048
#define MOE_D   1024
#define MOE_E   8
#define MOE_H   1024
#define MOE_HS  2048

// ---- fragment loaders -----------------------------------------------------
// A-frag (16x32 bf16, per ISA layout): lane = m + 16*half.
//   lanes 0-15 hold K[k..k+8) in v[0..7] and K[k+16..k+24) in v[8..15]
//   lanes 16-31 hold K[k+8..k+16) and K[k+24..k+32)
// => two contiguous 8-element chunks at (k + half*8) and (k + 16 + half*8).
// B-frag (32x16) has the identical per-lane packing with N in place of M, so
// B rows are simply rows of W for out = X * W^T. No transpose required.

__device__ __forceinline__ v16bf frag_from_f32(const float* __restrict__ p, int k, int half) {
  const v4f* q0 = (const v4f*)(p + k + half * 8);
  const v4f* q1 = (const v4f*)(p + k + 16 + half * 8);
  v4f a0 = q0[0], a1 = q0[1], b0 = q1[0], b1 = q1[1];
  v16bf r;
#pragma unroll
  for (int i = 0; i < 4; i++) {
    r[i]      = (__bf16)a0[i];
    r[4 + i]  = (__bf16)a1[i];
    r[8 + i]  = (__bf16)b0[i];
    r[12 + i] = (__bf16)b1[i];
  }
  return r;
}

__device__ __forceinline__ v16bf frag_from_bf16(const bf16_t* __restrict__ p, int k, int half) {
  v8bf c0 = *(const v8bf*)(p + k + half * 8);
  v8bf c1 = *(const v8bf*)(p + k + 16 + half * 8);
  v16bf r;
#pragma unroll
  for (int i = 0; i < 8; i++) { r[i] = c0[i]; r[8 + i] = c1[i]; }
  return r;
}

// ---- K0: zero control block (counts[8], cursors[8], offsets[9]) -----------
__global__ void moe_zero_ctrl(int* __restrict__ p) {
  if (threadIdx.x < 64) p[threadIdx.x] = 0;
}

// ---- K1: router (top-1) + bf16 copy of x ----------------------------------
// grid 256 x 256 threads; one wave per token.
__global__ __launch_bounds__(256)
void moe_router(const float* __restrict__ x, const float* __restrict__ rDE,
                int* __restrict__ eid, float* __restrict__ score,
                bf16_t* __restrict__ xt_bf, int* __restrict__ counts) {
  int wave = threadIdx.x >> 5, lane = threadIdx.x & 31;
  long long t = (long long)blockIdx.x * 8 + wave;
  const float* xr = x + t * MOE_D;
  float acc[8];
#pragma unroll
  for (int e = 0; e < 8; e++) acc[e] = 0.f;
#pragma unroll 4
  for (int j = 0; j < MOE_D / 32; j++) {
    int idx = lane + 32 * j;
    float xv = xr[idx];
    const v4f* rp = (const v4f*)(rDE + (long long)idx * MOE_E);
    v4f r0 = rp[0], r1 = rp[1];
    acc[0] += xv * r0[0]; acc[1] += xv * r0[1]; acc[2] += xv * r0[2]; acc[3] += xv * r0[3];
    acc[4] += xv * r1[0]; acc[5] += xv * r1[1]; acc[6] += xv * r1[2]; acc[7] += xv * r1[3];
    xt_bf[t * MOE_D + idx] = (bf16_t)xv;
  }
#pragma unroll
  for (int off = 16; off >= 1; off >>= 1)
#pragma unroll
    for (int e = 0; e < 8; e++) acc[e] += __shfl_xor(acc[e], off, 32);
  if (lane == 0) {
    int best = 0; float bv = acc[0];
#pragma unroll
    for (int e = 1; e < 8; e++)
      if (acc[e] > bv) { bv = acc[e]; best = e; }   // first-max wins, like argmax
    eid[t] = best;
    score[t] = 1.f / (1.f + __expf(-bv));
    atomicAdd(&counts[best], 1);
  }
}

// ---- K2: exclusive scan over 8 counts -------------------------------------
__global__ void moe_scan(const int* __restrict__ counts, int* __restrict__ offsets) {
  if (threadIdx.x == 0) {
    int s = 0;
    for (int e = 0; e < MOE_E; e++) { offsets[e] = s; s += counts[e]; }
    offsets[MOE_E] = s;
  }
}

// ---- K3: scatter tokens into expert buckets (writes bf16(x*score)) --------
__global__ __launch_bounds__(256)
void moe_scatter(const float* __restrict__ x, const int* __restrict__ eid,
                 const float* __restrict__ score, const int* __restrict__ offsets,
                 int* __restrict__ cursors, int* __restrict__ tok_of_slot,
                 bf16_t* __restrict__ xg) {
  __shared__ int sslot;
  long long t = blockIdx.x;
  if (threadIdx.x == 0) {
    int e = eid[t];
    int slot = offsets[e] + atomicAdd(&cursors[e], 1);
    tok_of_slot[slot] = (int)t;
    sslot = slot;
  }
  __syncthreads();
  long long slot = sslot;
  float s = score[t];
  const float* xr = x + t * MOE_D;
  bf16_t* dst = xg + slot * MOE_D;
  for (int i = threadIdx.x; i < MOE_D; i += 256) dst[i] = (bf16_t)(xr[i] * s);
}

// ---- K4/K6: fused SwiGLU up-projection:  H = silu(A W_g^T) * (A W_u^T) ----
// A: [Mtot, ldA] bf16 (row-major). W: gate rows [0,Hdim), up rows [Hdim,2*Hdim),
// each row of length K fp32 (converted to bf16 in-register).
// Block tile 128x128; 8 waves as 2x4; each wave: 64x32 = 4x2 tiles, gate+up.
__global__ __launch_bounds__(256)
void moe_swiglu_gemm(const bf16_t* __restrict__ A, const float* __restrict__ W,
                     bf16_t* __restrict__ Hout,
                     int K, int Hdim, int MT, int NT, int routed,
                     const int* __restrict__ offsets, long long w_stride,
                     int Mtot, int ldA) {
  int bid = blockIdx.x;
  int e = 0, tm, tn;
  long long bucket_start = 0; int count = Mtot;
  if (routed) {
    int per_e = MT * NT;
    e = bid / per_e; int rem = bid - e * per_e;
    tm = rem / NT; tn = rem - tm * NT;
    int b0 = offsets[e], b1 = offsets[e + 1];
    bucket_start = b0; count = b1 - b0;
    if (tm * 128 >= count) return;              // empty tile for this expert
  } else {
    tm = bid / NT; tn = bid - tm * NT;
  }
  const float* We = W + (long long)e * w_stride;

  int lane = threadIdx.x & 31, wave = threadIdx.x >> 5;
  int wm = wave >> 2, wn = wave & 3;            // 2 x 4 wave grid
  int mlane = lane & 15, half = lane >> 4;

  long long rowA[4];
#pragma unroll
  for (int s = 0; s < 4; s++) {
    long long r = bucket_start + tm * 128 + wm * 64 + s * 16 + mlane;
    rowA[s] = (r < Mtot) ? r : (long long)(Mtot - 1);   // clamp; masked at store
  }
  int colB[2];
#pragma unroll
  for (int t = 0; t < 2; t++) colB[t] = tn * 128 + wn * 32 + t * 16 + mlane;

  v8f zv = {0.f, 0.f, 0.f, 0.f, 0.f, 0.f, 0.f, 0.f};
  v8f accg[4][2], accu[4][2];
#pragma unroll
  for (int s = 0; s < 4; s++)
#pragma unroll
    for (int t = 0; t < 2; t++) { accg[s][t] = zv; accu[s][t] = zv; }

  for (int k = 0; k < K; k += 32) {
    v16bf af[4];
#pragma unroll
    for (int s = 0; s < 4; s++) af[s] = frag_from_bf16(A + rowA[s] * ldA, k, half);
    v16bf bg[2], bu[2];
#pragma unroll
    for (int t = 0; t < 2; t++) {
      bg[t] = frag_from_f32(We + (long long)colB[t] * K, k, half);
      bu[t] = frag_from_f32(We + (long long)(Hdim + colB[t]) * K, k, half);
    }
#pragma unroll
    for (int s = 0; s < 4; s++)
#pragma unroll
      for (int t = 0; t < 2; t++) {
        accg[s][t] = WMMA_BF16(af[s], bg[t], accg[s][t]);
        accu[s][t] = WMMA_BF16(af[s], bu[t], accu[s][t]);
      }
  }

  // epilogue: h = silu(g) * u, store bf16.
  // C/D layout: VGPR i -> row (i + 8*half) in tile, col = lane&15.
#pragma unroll
  for (int s = 0; s < 4; s++)
#pragma unroll
    for (int t = 0; t < 2; t++)
#pragma unroll
      for (int i = 0; i < 8; i++) {
        int rl = tm * 128 + wm * 64 + s * 16 + i + 8 * half;
        if (rl < count) {
          float g = accg[s][t][i], u = accu[s][t][i];
          float h = (g / (1.f + __expf(-g))) * u;
          long long row = bucket_start + rl;
          Hout[row * (long long)Hdim + colB[t]] = (bf16_t)h;
        }
      }
}

// ---- K5/K7: down-projection:  Out = H W2^T (+ scatter-add for routed) -----
__global__ __launch_bounds__(256)
void moe_down_gemm(const bf16_t* __restrict__ A, const float* __restrict__ W,
                   float* __restrict__ Out,
                   int K, int Dd, int MT, int NT, int routed,
                   const int* __restrict__ offsets, const int* __restrict__ tok_of_slot,
                   long long w_stride, int Mtot, int ldA) {
  int bid = blockIdx.x;
  int e = 0, tm, tn;
  long long bucket_start = 0; int count = Mtot;
  if (routed) {
    int per_e = MT * NT;
    e = bid / per_e; int rem = bid - e * per_e;
    tm = rem / NT; tn = rem - tm * NT;
    int b0 = offsets[e], b1 = offsets[e + 1];
    bucket_start = b0; count = b1 - b0;
    if (tm * 128 >= count) return;
  } else {
    tm = bid / NT; tn = bid - tm * NT;
  }
  const float* We = W + (long long)e * w_stride;

  int lane = threadIdx.x & 31, wave = threadIdx.x >> 5;
  int wm = wave >> 2, wn = wave & 3;
  int mlane = lane & 15, half = lane >> 4;

  long long rowA[4];
#pragma unroll
  for (int s = 0; s < 4; s++) {
    long long r = bucket_start + tm * 128 + wm * 64 + s * 16 + mlane;
    rowA[s] = (r < Mtot) ? r : (long long)(Mtot - 1);
  }
  int colB[2];
#pragma unroll
  for (int t = 0; t < 2; t++) colB[t] = tn * 128 + wn * 32 + t * 16 + mlane;

  v8f zv = {0.f, 0.f, 0.f, 0.f, 0.f, 0.f, 0.f, 0.f};
  v8f acc[4][2];
#pragma unroll
  for (int s = 0; s < 4; s++)
#pragma unroll
    for (int t = 0; t < 2; t++) acc[s][t] = zv;

  for (int k = 0; k < K; k += 32) {
    v16bf af[4];
#pragma unroll
    for (int s = 0; s < 4; s++) af[s] = frag_from_bf16(A + rowA[s] * ldA, k, half);
    v16bf bw[2];
#pragma unroll
    for (int t = 0; t < 2; t++) bw[t] = frag_from_f32(We + (long long)colB[t] * K, k, half);
#pragma unroll
    for (int s = 0; s < 4; s++)
#pragma unroll
      for (int t = 0; t < 2; t++) acc[s][t] = WMMA_BF16(af[s], bw[t], acc[s][t]);
  }

#pragma unroll
  for (int s = 0; s < 4; s++)
#pragma unroll
    for (int t = 0; t < 2; t++)
#pragma unroll
      for (int i = 0; i < 8; i++) {
        int rl = tm * 128 + wm * 64 + s * 16 + i + 8 * half;
        if (rl >= count) continue;
        float v = acc[s][t][i];
        if (routed) {
          // top-1 => each (token, col) owned by exactly one block: plain RMW.
          long long slot = bucket_start + rl;
          int tok = tok_of_slot[slot];
          float* o = Out + (long long)tok * Dd + colB[t];
          *o = *o + v;
        } else {
          Out[(bucket_start + rl) * (long long)Dd + colB[t]] = v;
        }
      }
}

// ---------------------------------------------------------------------------
extern "C" void kernel_launch(void* const* d_in, const int* in_sizes, int n_in,
                              void* d_out, int out_size, void* d_ws, size_t ws_size,
                              hipStream_t stream) {
  const float* x      = (const float*)d_in[0];   // [2,1024,1024]
  const float* router = (const float*)d_in[1];   // [1024, 8]
  const float* w13    = (const float*)d_in[2];   // [8, 2048, 1024]
  const float* w2     = (const float*)d_in[3];   // [8, 1024, 1024]
  const float* w13s   = (const float*)d_in[4];   // [4096, 1024]
  const float* w2s    = (const float*)d_in[5];   // [1024, 2048]
  float* out = (float*)d_out;                    // [2,1024,1024]

  char* ws = (char*)d_ws;
  // control block (256 B): counts @0, cursors @32, offsets @64
  int* counts  = (int*)(ws + 0);
  int* cursors = (int*)(ws + 32);
  int* offsets = (int*)(ws + 64);
  int*    eid   = (int*)  (ws + 256);                        // 8 KB
  float*  score = (float*)(ws + 256 + 8192);                 // 8 KB
  int*    tos   = (int*)  (ws + 256 + 16384);                // 8 KB
  bf16_t* xt    = (bf16_t*)(ws + 24832);                     // 4 MB  [N,D]
  bf16_t* xg    = (bf16_t*)(ws + 24832 + 4194304);           // 4 MB  [N,D] shuffled
  bf16_t* hb    = (bf16_t*)(ws + 24832 + 2 * 4194304);       // 4 MB  [N,H]
  bf16_t* hs    = (bf16_t*)(ws + 24832 + 3 * 4194304);       // 8 MB  [N,Hs]
  (void)in_sizes; (void)n_in; (void)out_size; (void)ws_size;

  moe_zero_ctrl<<<1, 64, 0, stream>>>((int*)ws);
  moe_router<<<MOE_N / 8, 256, 0, stream>>>(x, router, eid, score, xt, counts);
  moe_scan<<<1, 32, 0, stream>>>(counts, offsets);
  moe_scatter<<<MOE_N, 256, 0, stream>>>(x, eid, score, offsets, cursors, tos, xg);

  // routed SwiGLU: E(8) x MT(16) x NT(H/128=8) = 1024 blocks (early-exit).
  moe_swiglu_gemm<<<MOE_E * 16 * 8, 256, 0, stream>>>(
      xg, w13, hb, /*K=*/MOE_D, /*Hdim=*/MOE_H, /*MT=*/16, /*NT=*/8,
      /*routed=*/1, offsets, /*w_stride=*/(long long)2 * MOE_H * MOE_D,
      /*Mtot=*/MOE_N, /*ldA=*/MOE_D);

  // shared SwiGLU: MT(16) x NT(Hs/128=16) = 256 blocks.
  moe_swiglu_gemm<<<16 * 16, 256, 0, stream>>>(
      xt, w13s, hs, /*K=*/MOE_D, /*Hdim=*/MOE_HS, /*MT=*/16, /*NT=*/16,
      /*routed=*/0, nullptr, 0, /*Mtot=*/MOE_N, /*ldA=*/MOE_D);

  // shared down: writes Out fully (16 x 8 = 128 blocks).
  moe_down_gemm<<<16 * 8, 256, 0, stream>>>(
      hs, w2s, out, /*K=*/MOE_HS, /*Dd=*/MOE_D, /*MT=*/16, /*NT=*/8,
      /*routed=*/0, nullptr, nullptr, 0, /*Mtot=*/MOE_N, /*ldA=*/MOE_HS);

  // routed down: scatter-add into Out (after shared wrote it; same stream).
  moe_down_gemm<<<MOE_E * 16 * 8, 256, 0, stream>>>(
      hb, w2, out, /*K=*/MOE_H, /*Dd=*/MOE_D, /*MT=*/16, /*NT=*/8,
      /*routed=*/1, offsets, tos, /*w_stride=*/(long long)MOE_D * MOE_H,
      /*Mtot=*/MOE_N, /*ldA=*/MOE_H);
}
